// GraphConvModel_22677427323530
// MI455X (gfx1250) — matrix operand order repaired and checked by reading, not compiled, not run on hardware
//
#include <hip/hip_runtime.h>
#include <hip/hip_bf16.h>
#include <stdint.h>
#include <stddef.h>

// ---------------------------------------------------------------------------
// GCN forward for MI455X (gfx1250, wave32, WMMA).
// bf16 activations/weights + v_wmma_f32_16x16x32_bf16 (fp32 accumulate).
// B tile staged transposed in LDS -> ds_load_b128 fragment loads.
// Edge scatter via global_atomic_add_f32 (L2-resident: node matrix << 192MB).
// ---------------------------------------------------------------------------

typedef __attribute__((ext_vector_type(16))) __bf16 v16bf;
typedef __attribute__((ext_vector_type(8)))  float  v8f;

#define F_DIM 128

union Frag16 {            // 16 bf16 = 32 bytes = two b128 pieces
  uint4  q[2];
  v16bf  v;
};

__device__ __forceinline__ float tanhexp_f(float x) {
  // x * tanh(exp(x)); expf overflow -> inf, tanhf(inf) = 1 (safe)
  return x * tanhf(expf(x));
}

// ---------------- small utility kernels ----------------

__global__ void cvt_f32_bf16(const float* __restrict__ in, __bf16* __restrict__ out, int n) {
  int i = blockIdx.x * blockDim.x + threadIdx.x;
  if (i < n) out[i] = (__bf16)in[i];
}

__global__ void zero_f32(float* __restrict__ p, int n) {
  int i = blockIdx.x * blockDim.x + threadIdx.x;
  if (i < n) p[i] = 0.0f;
}

__global__ void deg_init(float* __restrict__ deg, int n) {
  int i = blockIdx.x * blockDim.x + threadIdx.x;
  if (i < n) deg[i] = 1.0f;   // self-loop contribution
}

__global__ void deg_accum(const int* __restrict__ dst, float* __restrict__ deg, int E) {
  int i = blockIdx.x * blockDim.x + threadIdx.x;
  if (i < E) atomicAdd(&deg[dst[i]], 1.0f);
}

__global__ void dinv_kernel(const float* __restrict__ deg, float* __restrict__ dinv, int n) {
  int i = blockIdx.x * blockDim.x + threadIdx.x;
  if (i < n) dinv[i] = rsqrtf(deg[i]);   // deg >= 1 always
}

// ---------------- WMMA GEMM: D_bf16[M,Nc] = act(A_bf16[M,K] @ B_bf16[K,Nc] + bias) ----------------
// One 16x16 output tile per wave; 4 waves per block share one B tile staged
// TRANSPOSED in LDS (Bs[n][k], padded stride) so each lane's B fragment is two
// contiguous 16B chunks -> ds_load_b128. A fragment likewise = two b128 loads.
// Requires: M % 16 == 0, K % 32 == 0, Nc % 16 == 0, K <= 256.

#define KMAX 256
#define KPAD 8   // pad stride so cooperative stores & fragment reads spread banks

template <int BIAS, int ACT>
__global__ __launch_bounds__(128)
void wmma_gemm_bf16(const __bf16* __restrict__ A, const __bf16* __restrict__ B,
                    const float* __restrict__ bias, __bf16* __restrict__ D,
                    int M, int K, int Nc) {
  __shared__ __align__(16) __bf16 Bs[16 * (KMAX + KPAD)];   // transposed tile

  const int lane  = threadIdx.x;                  // 0..31
  const int wv    = threadIdx.y;                  // 0..3
  const int nTile = blockIdx.x;
  const int mTile = blockIdx.y * 4 + wv;
  const int n0    = nTile * 16;
  const int m0    = mTile * 16;
  const int ldb   = K + KPAD;                     // LDS row stride (16B multiple *2)

  // Cooperative transpose-load of the shared B tile (all 128 threads).
  // Global reads coalesced over N; LDS stores stride ldb (banks spread by pad).
  const int tid = wv * 32 + lane;
  for (int idx = tid; idx < K * 16; idx += 128) {
    int kk = idx >> 4, nn = idx & 15;
    Bs[nn * ldb + kk] = B[(size_t)kk * Nc + (n0 + nn)];
  }
  __syncthreads();

  if (m0 < M) {                                   // wave-uniform: EXEC all-ones inside
    const int hf = lane >> 4;                     // lane half (0: lanes 0-15, 1: 16-31)
    const int lm = lane & 15;
    const int arow = m0 + lm;

    v8f c = {0.f, 0.f, 0.f, 0.f, 0.f, 0.f, 0.f, 0.f};

    for (int k0 = 0; k0 < K; k0 += 32) {
      // A 16x32 bf16 fragment: lane lm = row M; elements 0..7 -> K = k0+8*hf..+7,
      // elements 8..15 -> K = k0+16+8*hf..+7  (two contiguous 16B chunks)
      Frag16 fa;
      const __bf16* ap = A + (size_t)arow * K + k0 + 8 * hf;
      fa.q[0] = *reinterpret_cast<const uint4*>(ap);
      fa.q[1] = *reinterpret_cast<const uint4*>(ap + 16);

      // B 32x16 bf16 fragment: lane lm = col N; elements 0..7 -> K = k0+8*hf..+7,
      // elements 8..15 -> K = k0+16+8*hf..+7 (two 16B chunks) -> two ds_load_b128.
      Frag16 fb;
      const __bf16* bp = &Bs[lm * ldb + k0 + 8 * hf];
      fb.q[0] = *reinterpret_cast<const uint4*>(bp);
      fb.q[1] = *reinterpret_cast<const uint4*>(bp + 16);

      c = __builtin_amdgcn_wmma_f32_16x16x32_bf16(
              /*neg_a=*/false, fa.v, /*neg_b=*/false, fb.v,
              /*c_mod=*/(short)0, c, /*reuse_a=*/false, /*reuse_b=*/false);
    }

    // C/D 16x16 f32: element r -> row m0 + r + 8*hf, col n0 + lm
    const float bval = BIAS ? bias[n0 + lm] : 0.0f;
#pragma unroll
    for (int r = 0; r < 8; ++r) {
      int row = m0 + r + 8 * hf;
      float v = c[r] + bval;
      if (ACT) v = tanhexp_f(v);
      D[(size_t)row * Nc + (n0 + lm)] = (__bf16)v;
    }
  }
}

// ---------------- edge scatter: acc[dst] += t[src] * dinv[src]*dinv[dst] ----------------
// One wave32 per edge, each lane covers 4 features (b64 bf16 load + 4 f32 atomics).

__global__ __launch_bounds__(256)
void scatter_edges(const int* __restrict__ src, const int* __restrict__ dst,
                   const float* __restrict__ dinv, const __bf16* __restrict__ t,
                   float* __restrict__ acc, int E) {
  int e = blockIdx.x * 8 + (threadIdx.x >> 5);
  if (e >= E) return;
  int lane = threadIdx.x & 31;
  int s = src[e], d = dst[e];
  float nrm = dinv[s] * dinv[d];
  union { uint2 u; __bf16 h[4]; } cv;
  cv.u = *reinterpret_cast<const uint2*>(t + (size_t)s * F_DIM + lane * 4);
  float* ap = acc + (size_t)d * F_DIM + lane * 4;
#pragma unroll
  for (int j = 0; j < 4; ++j)
    atomicAdd(ap + j, (float)cv.h[j] * nrm);
}

// ---------------- finalize: h = tanhexp(acc + t*dinv^2 (self loop) + bias) -> bf16 ----------------

__global__ __launch_bounds__(F_DIM)
void finalize_nodes(const float* __restrict__ acc, const __bf16* __restrict__ t,
                    const float* __restrict__ dinv, const float* __restrict__ bias,
                    __bf16* __restrict__ h, int N) {
  int i = blockIdx.x;
  if (i >= N) return;
  int f = threadIdx.x;
  float di = dinv[i];
  float v = acc[(size_t)i * F_DIM + f]
          + (float)t[(size_t)i * F_DIM + f] * di * di
          + bias[f];
  h[(size_t)i * F_DIM + f] = (__bf16)tanhexp_f(v);
}

// ---------------- graph gather: g[seg[i]] += h[i] ----------------
// One wave32 per node, each lane covers 4 features.

__global__ __launch_bounds__(256)
void gather_nodes(const int* __restrict__ seg, const __bf16* __restrict__ h,
                  float* __restrict__ g, int N) {
  int i = blockIdx.x * 8 + (threadIdx.x >> 5);
  if (i >= N) return;
  int lane = threadIdx.x & 31;
  int sg = seg[i];
  union { uint2 u; __bf16 v[4]; } cv;
  cv.u = *reinterpret_cast<const uint2*>(h + (size_t)i * F_DIM + lane * 4);
  float* gp = g + (size_t)sg * F_DIM + lane * 4;
#pragma unroll
  for (int j = 0; j < 4; ++j)
    atomicAdd(gp + j, (float)cv.v[j]);
}

// ---------------- final 128 -> 1 head (scalar dot, tiny) ----------------

__global__ void out_head(const __bf16* __restrict__ m2, const float* __restrict__ W,
                         const float* __restrict__ b, float* __restrict__ out, int G) {
  int gi = blockIdx.x * blockDim.x + threadIdx.x;
  if (gi >= G) return;
  float s = b[0];
#pragma unroll 8
  for (int k = 0; k < F_DIM; ++k)
    s += (float)m2[(size_t)gi * F_DIM + k] * W[k];
  out[gi] = s;
}

// ---------------------------------------------------------------------------

extern "C" void kernel_launch(void* const* d_in, const int* in_sizes, int n_in,
                              void* d_out, int out_size, void* d_ws, size_t ws_size,
                              hipStream_t stream) {
  const float* x   = (const float*)d_in[0];
  const int*  edge = (const int*)d_in[1];
  const int*  seg  = (const int*)d_in[2];
  const float* Wg[3] = {(const float*)d_in[3], (const float*)d_in[5], (const float*)d_in[7]};
  const float* bg[3] = {(const float*)d_in[4], (const float*)d_in[6], (const float*)d_in[8]};
  const float* Wl0 = (const float*)d_in[9];   const float* bl0 = (const float*)d_in[10];
  const float* Wl1 = (const float*)d_in[11];  const float* bl1 = (const float*)d_in[12];
  const float* Wl2 = (const float*)d_in[13];  const float* bl2 = (const float*)d_in[14];

  const int N = in_sizes[0] / F_DIM;   // 100000
  const int E = in_sizes[1] / 2;       // 1600000
  const int G = out_size;              // 2048
  const int H1 = 256;                  // AFF_HID[0]

  const int* srcIdx = edge;            // edge_index[0]
  const int* dstIdx = edge + E;        // edge_index[1]

  // --- carve workspace (256B aligned) ---
  char* p = (char*)d_ws;
  auto take = [&](size_t bytes) -> void* {
    void* r = (void*)p;
    p += (bytes + 255) & ~(size_t)255;
    return r;
  };
  float*  acc  = (float*) take((size_t)N * F_DIM * 4);   // scatter accumulator
  __bf16* hb   = (__bf16*)take((size_t)N * F_DIM * 2);   // layer input (bf16)
  __bf16* tb   = (__bf16*)take((size_t)N * F_DIM * 2);   // h @ W (bf16)
  float*  deg  = (float*) take((size_t)N * 4);
  float*  dinv = (float*) take((size_t)N * 4);
  __bf16* wb   = (__bf16*)take((size_t)256 * 256 * 2);   // weight staging (bf16)
  float*  gf   = (float*) take((size_t)G * F_DIM * 4);   // graph gather accumulator
  __bf16* gb   = (__bf16*)take((size_t)G * F_DIM * 2);
  __bf16* m1   = (__bf16*)take((size_t)G * H1   * 2);
  __bf16* m2   = (__bf16*)take((size_t)G * F_DIM * 2);

  // --- degrees + normalization ---
  deg_init   <<<(N + 255) / 256, 256, 0, stream>>>(deg, N);
  deg_accum  <<<(E + 255) / 256, 256, 0, stream>>>(dstIdx, deg, E);
  dinv_kernel<<<(N + 255) / 256, 256, 0, stream>>>(deg, dinv, N);

  // --- x -> bf16 ---
  cvt_f32_bf16<<<((N * F_DIM) + 255) / 256, 256, 0, stream>>>(x, hb, N * F_DIM);

  // --- 3 GCN layers ---
  for (int l = 0; l < 3; ++l) {
    cvt_f32_bf16<<<(F_DIM * F_DIM + 255) / 256, 256, 0, stream>>>(Wg[l], wb, F_DIM * F_DIM);
    {
      dim3 gg(F_DIM / 16, (N / 16 + 3) / 4);
      dim3 bb(32, 4);
      wmma_gemm_bf16<0, 0><<<gg, bb, 0, stream>>>(hb, wb, nullptr, tb, N, F_DIM, F_DIM);
    }
    zero_f32<<<((N * F_DIM) + 255) / 256, 256, 0, stream>>>(acc, N * F_DIM);
    scatter_edges <<<(E + 7) / 8, 256, 0, stream>>>(srcIdx, dstIdx, dinv, tb, acc, E);
    finalize_nodes<<<N, F_DIM, 0, stream>>>(acc, tb, dinv, bg[l], hb, N);
  }

  // --- graph gather ---
  zero_f32    <<<(G * F_DIM + 255) / 256, 256, 0, stream>>>(gf, G * F_DIM);
  gather_nodes<<<(N + 7) / 8, 256, 0, stream>>>(seg, hb, gf, N);
  cvt_f32_bf16<<<(G * F_DIM + 255) / 256, 256, 0, stream>>>(gf, gb, G * F_DIM);

  // --- MLP head: 128 -> 256 -> 128 -> 1 ---
  cvt_f32_bf16<<<(F_DIM * H1 + 255) / 256, 256, 0, stream>>>(Wl0, wb, F_DIM * H1);
  {
    dim3 gg(H1 / 16, (G / 16 + 3) / 4);
    dim3 bb(32, 4);
    wmma_gemm_bf16<1, 1><<<gg, bb, 0, stream>>>(gb, wb, bl0, m1, G, F_DIM, H1);
  }
  cvt_f32_bf16<<<(H1 * F_DIM + 255) / 256, 256, 0, stream>>>(Wl1, wb, H1 * F_DIM);
  {
    dim3 gg(F_DIM / 16, (G / 16 + 3) / 4);
    dim3 bb(32, 4);
    wmma_gemm_bf16<1, 1><<<gg, bb, 0, stream>>>(m1, wb, bl1, m2, G, H1, F_DIM);
  }
  out_head<<<(G + 255) / 256, 256, 0, stream>>>(m2, Wl2, bl2, (float*)d_out, G);
}
